// GPTBlock_80719615361632
// MI455X (gfx1250) — compile-verified, matrix operand order checked by reading, not run on hardware
//
#include <hip/hip_runtime.h>
#include <hip/hip_bf16.h>

// ---------------------------------------------------------------------------
// GPT block (pre-LN attn + SwiGLU) for gfx1250.
// B=2 S=2048 D=2048 H=16 HD=128 FF=8192.
// ~0.58 TFLOP of GEMMs vs ~268MB of f32 weights (~11.5us @ 23.3TB/s):
// compute-bound -> run all matmuls on v_wmma_f32_16x16x32_f16 with on-the-fly
// f32->f16 conversion (no extra HBM traffic: f32 source must be read anyway).
// A-tiles (already f16) are staged to LDS with GLOBAL_LOAD_ASYNC_TO_LDS_B128
// (ASYNCcnt path), overlapping with the W f32->f16 convert done through VGPRs.
// ---------------------------------------------------------------------------

typedef __attribute__((ext_vector_type(16))) _Float16 v16h;
typedef __attribute__((ext_vector_type(8)))  _Float16 v8h;
typedef __attribute__((ext_vector_type(4)))  _Float16 v4h;
typedef __attribute__((ext_vector_type(8)))  float    v8f;
typedef __attribute__((ext_vector_type(4)))  float    v4f;

#define Bb  2
#define Ss  2048
#define Dd  2048
#define Hh  16
#define HDd 128
#define FFf 8192
#define Mm  (Bb*Ss)          // 4096 rows of the token matrix

// Build a 16x32 f16 A/B fragment (wave32 layout, CDNA5 ISA):
// lane = {m|n = lane&15, half = lane>>4}; element e<8 -> K=8*half+e,
// e>=8 -> K=16+8*half+(e-8). So per lane: two contiguous 8-half (16B) chunks.
__device__ __forceinline__ v16h frag16(const _Float16* p, int half) {
  v8h a = *(const v8h*)(p + 8 * half);
  v8h b = *(const v8h*)(p + 16 + 8 * half);
  return __builtin_shufflevector(a, b, 0,1,2,3,4,5,6,7,8,9,10,11,12,13,14,15);
}

__device__ __forceinline__ v8f wmma16(v16h a, v16h b, v8f c) {
  return __builtin_amdgcn_wmma_f32_16x16x32_f16(false, a, false, b,
                                                (short)0, c, false, false);
}

// Async global->LDS 16B copy (per active lane). ISA aperture rule: the LDS
// byte address is the low 32 bits of the flat shared pointer.
__device__ __forceinline__ void async_copy_b128(const _Float16* gsrc,
                                                _Float16* lds_dst) {
  unsigned lds_addr = (unsigned)(uintptr_t)lds_dst;
  asm volatile("global_load_async_to_lds_b128 %0, %1, off"
               :
               : "v"(lds_addr), "v"((unsigned long long)(uintptr_t)gsrc)
               : "memory");
}

__device__ __forceinline__ void wait_asynccnt0() {
  asm volatile("s_wait_asynccnt 0x0" ::: "memory");
}

// ---------------------------------------------------------------------------
// LayerNorm: one 256-thread block per row (D=2048 -> 8 floats/thread),
// emits f16 activations for the WMMA GEMMs.
// ---------------------------------------------------------------------------
__global__ __launch_bounds__(256)
void layernorm_f16(const float* __restrict__ x, const float* __restrict__ w,
                   const float* __restrict__ bv, _Float16* __restrict__ out) {
  __shared__ float red[16];
  const int tid = threadIdx.x;
  const size_t base = (size_t)blockIdx.x * Dd + (size_t)tid * 8;
  v4f a = *(const v4f*)(x + base);
  v4f b = *(const v4f*)(x + base + 4);

  float s = a[0]+a[1]+a[2]+a[3] + b[0]+b[1]+b[2]+b[3];
#pragma unroll
  for (int off = 16; off >= 1; off >>= 1) s += __shfl_xor(s, off);
  if ((tid & 31) == 0) red[tid >> 5] = s;
  __syncthreads();
  float tot = 0.f;
#pragma unroll
  for (int i = 0; i < 8; i++) tot += red[i];
  const float mu = tot * (1.0f / Dd);

  float vs = 0.f;
#pragma unroll
  for (int i = 0; i < 4; i++) { float d0 = a[i]-mu, d1 = b[i]-mu; vs += d0*d0 + d1*d1; }
#pragma unroll
  for (int off = 16; off >= 1; off >>= 1) vs += __shfl_xor(vs, off);
  if ((tid & 31) == 0) red[8 + (tid >> 5)] = vs;
  __syncthreads();
  float vtot = 0.f;
#pragma unroll
  for (int i = 0; i < 8; i++) vtot += red[8 + i];
  const float inv = rsqrtf(vtot * (1.0f / Dd) + 1e-5f);

  const int c = tid * 8;
  v8h o;
#pragma unroll
  for (int i = 0; i < 4; i++) {
    o[i]     = (_Float16)((a[i] - mu) * inv * w[c + i]     + bv[c + i]);
    o[i + 4] = (_Float16)((b[i] - mu) * inv * w[c + 4 + i] + bv[c + 4 + i]);
  }
  *(v8h*)(out + base) = o;
}

// ---------------------------------------------------------------------------
// Generic WMMA GEMM: O[M,N] = epi(A[M,K](f16) @ W[N,K](f32)^T + bias).
// 256 threads = 8 waves; 128x128 macro tile; wave tile 64x32 (4x2 accums);
// K-step 32. A staged to LDS via async b128 copies; W converted f32->f16 via
// VGPRs. EPI: 0 = f16 out, 1 = f32 out + residual add, 2 = silu(extra)*acc.
// ---------------------------------------------------------------------------
template <int EPI>
__global__ __launch_bounds__(256)
void gemm_wmma(const _Float16* __restrict__ A, const float* __restrict__ W,
               const float* __restrict__ bias, const float* __restrict__ resF,
               const _Float16* __restrict__ extraH, float* __restrict__ outF,
               _Float16* __restrict__ outH, int M, int N, int K) {
  __shared__ __attribute__((aligned(16))) _Float16 As[128][40];
  __shared__ __attribute__((aligned(16))) _Float16 Bs[128][40];

  const int tid  = threadIdx.x;
  const int lane = tid & 31;
  const int wave = tid >> 5;
  const int nTilesN = N >> 7;
  const int bm = blockIdx.x / nTilesN;
  const int bn = blockIdx.x % nTilesN;
  const size_t rowBase = (size_t)bm << 7;
  const size_t colBase = (size_t)bn << 7;

  const int wr = wave >> 2;     // 0..1  : 64-row band
  const int wc = wave & 3;      // 0..3  : 32-col band
  const int half = lane >> 4;
  const int l15  = lane & 15;

  v8f acc[4][2] = {};

  // staging maps
  const int arow = tid >> 2;          // 0..63 (x2 passes)
  const int acol = (tid & 3) * 8;     // 0,8,16,24 halves
  const int wrow = tid >> 3;          // 0..31 (x4 passes)
  const int wcol = (tid & 7) * 4;     // 0..28 floats

  for (int k0 = 0; k0 < K; k0 += 32) {
    __syncthreads();  // previous iteration's LDS readers are done
    // ---- stage A tile: async global->LDS 128-bit copies (ASYNCcnt) ----
#pragma unroll
    for (int i = 0; i < 2; i++) {
      const int r = arow + 64 * i;
      async_copy_b128(A + (rowBase + r) * (size_t)K + k0 + acol, &As[r][acol]);
    }
    // ---- stage W tile (f32 -> f16 -> LDS through VGPRs) ----
#pragma unroll
    for (int i = 0; i < 4; i++) {
      const int n = wrow + 32 * i;
      v4f wv = *(const v4f*)(W + (colBase + n) * (size_t)K + k0 + wcol);
      v4h hv;
#pragma unroll
      for (int q = 0; q < 4; q++) hv[q] = (_Float16)wv[q];
      *(v4h*)&Bs[n][wcol] = hv;
    }
    if (k0 + 32 < K)  // hint the next W tile toward GL2 (global_prefetch_b8)
      __builtin_prefetch(
          (const void*)(W + (colBase + (size_t)(tid >> 1)) * (size_t)K + k0 + 32), 0, 1);
    wait_asynccnt0();  // A-tile landed in LDS
    __syncthreads();

    // ---- compute: 4x2 sub-tiles of 16x16 per wave ----
    v16h af[4], bf[2];
#pragma unroll
    for (int i = 0; i < 4; i++)
      af[i] = frag16(&As[wr * 64 + i * 16 + l15][0], half);
#pragma unroll
    for (int j = 0; j < 2; j++)
      bf[j] = frag16(&Bs[wc * 32 + j * 16 + l15][0], half);
#pragma unroll
    for (int i = 0; i < 4; i++)
#pragma unroll
      for (int j = 0; j < 2; j++)
        acc[i][j] = wmma16(af[i], bf[j], acc[i][j]);
  }

  // ---- epilogue (C layout: row = r + 8*half, col = lane&15) ----
#pragma unroll
  for (int i = 0; i < 4; i++) {
#pragma unroll
    for (int j = 0; j < 2; j++) {
      const size_t col = colBase + wc * 32 + j * 16 + l15;
      const float bb = bias[col];
#pragma unroll
      for (int r = 0; r < 8; r++) {
        const size_t row = rowBase + wr * 64 + i * 16 + r + 8 * half;
        const size_t idx = row * (size_t)N + col;
        const float v = acc[i][j][r] + bb;
        if (EPI == 0) {
          outH[idx] = (_Float16)v;
        } else if (EPI == 1) {
          outF[idx] = v + resF[idx];
        } else {
          const float g = (float)extraH[idx];
          const float sg = g / (1.0f + __expf(-g));
          outH[idx] = (_Float16)(sg * v);
        }
      }
    }
  }
}

// ---------------------------------------------------------------------------
// V transpose: vT[b][h][hd][s] = qkv[(b*S+s)][2D + h*HD + hd] so P@V B-frags
// are contiguous along the key axis (two b128 global loads per fragment).
// ---------------------------------------------------------------------------
__global__ __launch_bounds__(256)
void transpose_v(const _Float16* __restrict__ qkv, _Float16* __restrict__ vT) {
  const size_t i = (size_t)blockIdx.x * 256 + threadIdx.x;  // over B*H*HD*S
  const int s  = (int)(i & (Ss - 1));
  size_t rest  = i >> 11;
  const int hd = (int)(rest & (HDd - 1)); rest >>= 7;
  const int h  = (int)(rest & (Hh - 1));
  const int b  = (int)(rest >> 4);
  vT[i] = qkv[((size_t)(b * Ss + s)) * (3 * Dd) + 2 * Dd + h * HDd + hd];
}

// ---------------------------------------------------------------------------
// Causal flash attention, one wave per 16-row Q tile (HD=128).
// Scores: 4 WMMAs per 16-key tile (Q,K frags straight from global — rows are
// K-contiguous). Online softmax with 16-lane shuffle reductions. P is pushed
// through a per-wave LDS tile to re-layout C->A, then 8 WMMAs of P(16x32) @
// V(32x128) per 32-key step against the pre-transposed Vt.
// ---------------------------------------------------------------------------
__global__ __launch_bounds__(256)
void flash_attn(const _Float16* __restrict__ qkv, const _Float16* __restrict__ vT,
                _Float16* __restrict__ ctx) {
  __shared__ __attribute__((aligned(16))) _Float16 P[8][16][40];

  const int lane = threadIdx.x & 31;
  const int wave = threadIdx.x >> 5;
  const int wt = blockIdx.x * 8 + wave;     // 4096 wave-tiles total
  const int qt = wt & 127;                  // S/16 = 128 q tiles
  const int bh = wt >> 7;
  const int b = bh >> 4, h = bh & 15;
  const int q0 = qt * 16;
  const int half = lane >> 4, l15 = lane & 15;

  const _Float16* Qb = qkv + (size_t)(b * Ss) * (3 * Dd) + h * HDd;
  const _Float16* Kb = Qb + Dd;
  const _Float16* Vt = vT + ((size_t)(b * Hh + h) * HDd) * Ss;

  v16h qf[4];
  {
    const _Float16* qrow = Qb + (size_t)(q0 + l15) * (3 * Dd);
#pragma unroll
    for (int c = 0; c < 4; c++) qf[c] = frag16(qrow + c * 32, half);
  }

  v8f cacc[8] = {};
  float mrow[8], lrow[8];
#pragma unroll
  for (int r = 0; r < 8; r++) { mrow[r] = -3.0e38f; lrow[r] = 0.f; }
  const float scale = 0.08838834764831845f;  // 1/sqrt(128)

  const int nkv = q0 + 16;
  for (int kv = 0; kv < nkv; kv += 32) {
    // ---- two 16x16 score tiles (keys kv..kv+31) ----
    v8f s0 = {}, s1 = {};
#pragma unroll
    for (int c = 0; c < 4; c++) {
      v16h kf = frag16(Kb + (size_t)(kv + l15) * (3 * Dd) + c * 32, half);
      s0 = wmma16(qf[c], kf, s0);
    }
#pragma unroll
    for (int c = 0; c < 4; c++) {
      v16h kf = frag16(Kb + (size_t)(kv + 16 + l15) * (3 * Dd) + c * 32, half);
      s1 = wmma16(qf[c], kf, s1);
    }

    // ---- online softmax per row (row = r + 8*half, col = lane&15) ----
#pragma unroll
    for (int r = 0; r < 8; r++) {
      const int qi = q0 + r + 8 * half;
      float a0 = (kv + l15      <= qi) ? s0[r] * scale : -3.0e38f;
      float a1 = (kv + 16 + l15 <= qi) ? s1[r] * scale : -3.0e38f;
      float mx = fmaxf(a0, a1);
#pragma unroll
      for (int off = 8; off >= 1; off >>= 1) mx = fmaxf(mx, __shfl_xor(mx, off));
      const float mnew = fmaxf(mrow[r], mx);
      const float corr = __expf(mrow[r] - mnew);
      const float p0 = __expf(a0 - mnew);
      const float p1 = __expf(a1 - mnew);
      float ps = p0 + p1;
#pragma unroll
      for (int off = 8; off >= 1; off >>= 1) ps += __shfl_xor(ps, off);
      lrow[r] = lrow[r] * corr + ps;
      mrow[r] = mnew;
#pragma unroll
      for (int nt = 0; nt < 8; nt++) cacc[nt][r] *= corr;
      P[wave][r + 8 * half][l15]      = (_Float16)p0;
      P[wave][r + 8 * half][16 + l15] = (_Float16)p1;
    }

    // ---- ctx += P(16x32) @ V(32x128); DS ops are in-order within a wave ----
    v16h pf = frag16(&P[wave][l15][0], half);
#pragma unroll
    for (int nt = 0; nt < 8; nt++) {
      v16h vf = frag16(Vt + (size_t)(nt * 16 + l15) * Ss + kv, half);
      cacc[nt] = wmma16(pf, vf, cacc[nt]);
    }
  }

  // ---- normalize and write ctx (f16, [B*S, D]) ----
#pragma unroll
  for (int r = 0; r < 8; r++) {
    const float inv = 1.0f / lrow[r];
    const size_t row = (size_t)b * Ss + q0 + r + 8 * half;
#pragma unroll
    for (int nt = 0; nt < 8; nt++) {
      const int col = h * HDd + nt * 16 + l15;
      ctx[row * Dd + col] = (_Float16)(cacc[nt][r] * inv);
    }
  }
}

// ---------------------------------------------------------------------------
// Orchestration
// ---------------------------------------------------------------------------
extern "C" void kernel_launch(void* const* d_in, const int* in_sizes, int n_in,
                              void* d_out, int out_size, void* d_ws, size_t ws_size,
                              hipStream_t stream) {
  (void)in_sizes; (void)n_in; (void)out_size; (void)ws_size;

  const float* x      = (const float*)d_in[0];
  const float* ln1_w  = (const float*)d_in[1];
  const float* ln1_b  = (const float*)d_in[2];
  const float* qkv_w  = (const float*)d_in[3];
  const float* qkv_b  = (const float*)d_in[4];
  const float* out_w  = (const float*)d_in[5];
  const float* out_b  = (const float*)d_in[6];
  const float* ln2_w  = (const float*)d_in[7];
  const float* ln2_b  = (const float*)d_in[8];
  const float* fc1_w  = (const float*)d_in[9];
  const float* fc1_b  = (const float*)d_in[10];
  const float* fc2_w  = (const float*)d_in[11];
  const float* fc2_b  = (const float*)d_in[12];
  const float* fc3_w  = (const float*)d_in[13];
  const float* fc3_b  = (const float*)d_in[14];

  // workspace layout (bytes); gate16 aliases qkv16+vT16 (both dead post-attn)
  char* ws = (char*)d_ws;
  _Float16* h16    = (_Float16*)(ws);                 // 4096*2048*2 = 16,777,216
  _Float16* qkv16  = (_Float16*)(ws + 16777216);      // 4096*6144*2 = 50,331,648
  _Float16* vT16   = (_Float16*)(ws + 67108864);      // 4096*2048*2 = 16,777,216
  _Float16* gate16 = (_Float16*)(ws + 16777216);      // 4096*8192*2 = 67,108,864 (alias)
  _Float16* ctx16  = (_Float16*)(ws + 83886080);      // 16,777,216
  float*    x1     = (float*)   (ws + 100663296);     // 4096*2048*4 = 33,554,432
  _Float16* gv16   = (_Float16*)(ws + 134217728);     // 67,108,864  (end 201,326,592)
  float*    xout   = (float*)d_out;

  // 1) LN1 -> h16 (f16)
  layernorm_f16<<<Mm, 256, 0, stream>>>(x, ln1_w, ln1_b, h16);
  // 2) QKV projection: [4096,6144] = h16 @ qkv_w^T + b   (f16 out)
  gemm_wmma<0><<<(Mm/128)*(3*Dd/128), 256, 0, stream>>>(
      h16, qkv_w, qkv_b, nullptr, nullptr, nullptr, qkv16, Mm, 3*Dd, Dd);
  // 3) Vt for the P@V fragments
  transpose_v<<<(Bb*Ss*Dd)/256, 256, 0, stream>>>(qkv16, vT16);
  // 4) causal flash attention -> ctx16 (f16)
  flash_attn<<<(Bb*Hh*(Ss/16))/8, 256, 0, stream>>>(qkv16, vT16, ctx16);
  // 5) out projection + residual: x1 = x + ctx @ out_w^T + b  (f32)
  gemm_wmma<1><<<(Mm/128)*(Dd/128), 256, 0, stream>>>(
      ctx16, out_w, out_b, x, nullptr, x1, nullptr, Mm, Dd, Dd);
  // 6) LN2 -> h16 reused (f16)
  layernorm_f16<<<Mm, 256, 0, stream>>>(x1, ln2_w, ln2_b, h16);
  // 7) fc1 (gate pre-activation, f16)
  gemm_wmma<0><<<(Mm/128)*(FFf/128), 256, 0, stream>>>(
      h16, fc1_w, fc1_b, nullptr, nullptr, nullptr, gate16, Mm, FFf, Dd);
  // 8) fc2 fused with silu(gate)*val -> gv16 (f16)
  gemm_wmma<2><<<(Mm/128)*(FFf/128), 256, 0, stream>>>(
      h16, fc2_w, fc2_b, nullptr, gate16, nullptr, gv16, Mm, FFf, Dd);
  // 9) fc3 + residual -> d_out (f32)
  gemm_wmma<1><<<(Mm/128)*(Dd/128), 256, 0, stream>>>(
      gv16, fc3_w, fc3_b, x1, nullptr, xout, nullptr, Mm, Dd, FFf);
}
